// Phi3Attention_65721589563822
// MI455X (gfx1250) — matrix operand order repaired and checked by reading, compiled, not run on hardware
//
#include <hip/hip_runtime.h>
#include <hip/hip_bf16.h>
#include <cstdint>

// Phi-3 attention block for MI455X (gfx1250, wave32, WMMA + TDM).
// Pipeline: cast->bf16, TDM-fed WMMA QKV GEMM, SuRoPE+repack (V transposed),
// causal flash-attention (wave per 16-row q tile, bf16 WMMA, fp32 softmax),
// TDM-fed WMMA output projection.

#define LSEQ   2048
#define DMODEL 3072
#define NHEAD  32
#define HDIM   96
#define OPROJ  (NHEAD * HDIM)   // 3072
#define OPQKV  9216             // H*HD + 2*KV*HD

#define ATTN_SCALE 0.10206207261596577f   // 96^-0.5
#define ROPE_SCALE 1.1902380714238083f    // sqrt(1 + ln(32)/ln(4096)) = sqrt(17/12)
#define THETA      10000.0f
#define NEG_BIG    (-1.0e30f)

typedef __attribute__((ext_vector_type(16))) __bf16 v16bf;
typedef __attribute__((ext_vector_type(8)))  float  v8f;
typedef __attribute__((ext_vector_type(4)))  unsigned int v4u;
typedef __attribute__((ext_vector_type(8)))  int v8i;
typedef __attribute__((ext_vector_type(4)))  int v4i;

union FragBF { v16bf v; uint4 u[2]; };

// A operand (16x32 bf16, M x K), source row-major [M][ldk].
// Lane (hf = lane>>4, m = lane&15) holds row m:
//   v0..v3 = K = hf*8 + 0..7 ; v4..v7 = K = 16 + hf*8 + 0..7  (2 bf16/VGPR)
__device__ __forceinline__ v16bf load_a_frag(const __bf16* base, int ldk, int lane) {
  const int hf = lane >> 4, m = lane & 15;
  const __bf16* p = base + (size_t)m * ldk + hf * 8;
  FragBF f;
  f.u[0] = *reinterpret_cast<const uint4*>(p);
  f.u[1] = *reinterpret_cast<const uint4*>(p + 16);
  return f.v;
}

// B operand (32x16 bf16, K x N), source stored as B^T row-major [N][ldk] (K contiguous).
// Lane (hf, n) holds column n, K = hf*8 + 0..7 and 16 + hf*8 + 0..7 (matches A).
__device__ __forceinline__ v16bf load_b_frag(const __bf16* base, int ldk, int lane) {
  const int hf = lane >> 4, nn = lane & 15;
  const __bf16* p = base + (size_t)nn * ldk + hf * 8;
  FragBF f;
  f.u[0] = *reinterpret_cast<const uint4*>(p);
  f.u[1] = *reinterpret_cast<const uint4*>(p + 16);
  return f.v;
}

// -------------------------------------------------------------------------
// TDM: DMA a rows x 64 bf16 tile (row stride = stride_elems) from global into
// LDS at byte offset lds_off. Pad 4 DWORDs (16B) after every 32 DWORDs
// (128B = one 64-elem row) -> LDS row stride 144B (72 bf16), which spreads
// the per-lane fragment reads across banks.
// D# packing per CDNA5 ISA ch.8 (groups 0/1; 2D tensor -> groups 2/3 zero).
// -------------------------------------------------------------------------
__device__ __forceinline__ void tdm_load_tile64w(
    unsigned lds_off, const __bf16* gsrc, unsigned stride_elems, unsigned rows) {
  const unsigned long long ga = (unsigned long long)(uintptr_t)gsrc;
  v4u g0;
  g0[0] = 1u;                                       // count=1, user descriptor
  g0[1] = lds_off;                                  // lds_addr (bytes)
  g0[2] = (unsigned)(ga & 0xFFFFFFFFull);           // global_addr[31:0]
  g0[3] = (unsigned)((ga >> 32) & 0x1FFFFFFull)     // global_addr[56:32]
          | 0x80000000u;                            // type=2 ("image")
  // data_size=1 (2B) | pad_enable | pad_interval=4 (32 DW) | pad_amount=3 (4 DW)
  const unsigned w0 = (1u << 16) | (1u << 20) | (4u << 22) | (3u << 25);
  v8i g1;
  g1[0] = (int)w0;
  g1[1] = (int)((stride_elems & 0xFFFFu) << 16);    // tensor_dim0[15:0]
  g1[2] = (int)(((stride_elems >> 16) & 0xFFFFu)    // tensor_dim0[31:16]
          | ((rows & 0xFFFFu) << 16));              // tensor_dim1[15:0]
  g1[3] = (int)(64u << 16);                         // tensor_dim1 hi=0 | tile_dim0=64
  g1[4] = (int)(rows & 0xFFFFu);                    // tile_dim1=rows | tile_dim2=0
  g1[5] = (int)stride_elems;                        // tensor_dim0_stride[31:0]
  g1[6] = 0;                                        // stride[47:32] | dim1_stride lo
  g1[7] = 0;
  const v4i z4 = {0, 0, 0, 0};
#if __clang_major__ >= 23
  const v8i z8 = {0, 0, 0, 0, 0, 0, 0, 0};
  __builtin_amdgcn_tensor_load_to_lds(g0, g1, z4, z4, z8, 0);
#else
  __builtin_amdgcn_tensor_load_to_lds(g0, g1, z4, z4, 0);
#endif
}

__global__ void cast_f32_bf16(const float* __restrict__ in, __bf16* __restrict__ out, long n) {
  long i = (long)blockIdx.x * blockDim.x + threadIdx.x;
  const long stride = (long)gridDim.x * blockDim.x;
  for (; i < n; i += stride) out[i] = (__bf16)in[i];
}

// C[M][N] = A[M][K] * B[N][K]^T, bf16 in / fp32 out.
// Block = 4 waves -> 128(M) x 64(N) tile, K stepped by 64; wave = 32x64.
// Wave 0 drives the Tensor Data Mover: double-buffered A/B tiles in LDS,
// s_wait_tensorcnt for pipeline depth 1; all waves consume via ds_load + wmma.
__global__ __launch_bounds__(128) void gemm_bf16_f32(
    const __bf16* __restrict__ A, const __bf16* __restrict__ B,
    float* __restrict__ C, int M, int N, int K) {
  __shared__ __align__(16) __bf16 Ab[2][128 * 72];  // 144B LDS row stride
  __shared__ __align__(16) __bf16 Bb[2][64 * 72];
  const int lane = threadIdx.x & 31;
  const int wave = threadIdx.x >> 5;
  const int m0 = blockIdx.y * 128;
  const int n0 = blockIdx.x * 64;
  const int hf = lane >> 4, nn = lane & 15;
  (void)M;

  const unsigned aoff = (unsigned)(uintptr_t)(&Ab[0][0]);
  const unsigned boff = (unsigned)(uintptr_t)(&Bb[0][0]);
  const unsigned ABUF = 128u * 72u * 2u;            // bytes per A buffer
  const unsigned BBUF = 64u * 72u * 2u;             // bytes per B buffer

  const int nsteps = K >> 6;                        // K / 64
  if (wave == 0) {
    tdm_load_tile64w(aoff, A + (size_t)m0 * K, (unsigned)K, 128u);
    tdm_load_tile64w(boff, B + (size_t)n0 * K, (unsigned)K, 64u);
  }

  v8f acc0[4] = {}, acc1[4] = {};
  for (int i = 0; i < nsteps; ++i) {
    const int cur = i & 1;
    if (i + 1 < nsteps) {
      if (wave == 0) {
        const int k2 = (i + 1) << 6;
        tdm_load_tile64w(aoff + (unsigned)(cur ^ 1) * ABUF,
                         A + (size_t)m0 * K + k2, (unsigned)K, 128u);
        tdm_load_tile64w(boff + (unsigned)(cur ^ 1) * BBUF,
                         B + (size_t)n0 * K + k2, (unsigned)K, 64u);
      }
      __builtin_amdgcn_s_wait_tensorcnt(2);         // current step's 2 DMAs done
    } else {
      __builtin_amdgcn_s_wait_tensorcnt(0);
    }
    __syncthreads();                                // LDS tiles visible to all waves

    const __bf16* As = &Ab[cur][0] + (size_t)(wave * 32) * 72;
    const __bf16* Bs = &Bb[cur][0];
#pragma unroll
    for (int kk = 0; kk < 2; ++kk) {
      // Batch all fragment loads first so the ds_load clause overlaps the wmmas.
      v16bf a0 = load_a_frag(As + kk * 32, 72, lane);
      v16bf a1 = load_a_frag(As + (size_t)16 * 72 + kk * 32, 72, lane);
      v16bf b[4];
#pragma unroll
      for (int t = 0; t < 4; ++t)
        b[t] = load_b_frag(Bs + (size_t)(t * 16) * 72 + kk * 32, 72, lane);
#pragma unroll
      for (int t = 0; t < 4; ++t) {
        acc0[t] = __builtin_amdgcn_wmma_f32_16x16x32_bf16(
            false, a0, false, b[t], (short)0, acc0[t], false, false);
        acc1[t] = __builtin_amdgcn_wmma_f32_16x16x32_bf16(
            false, a1, false, b[t], (short)0, acc1[t], false, false);
      }
    }
    __syncthreads();                                // done consuming before overwrite
  }

  const int mw = m0 + wave * 32;
#pragma unroll
  for (int t = 0; t < 4; ++t)
#pragma unroll
    for (int r = 0; r < 8; ++r) {
      C[(size_t)(mw + hf * 8 + r) * N + n0 + t * 16 + nn]      = acc0[t][r];
      C[(size_t)(mw + 16 + hf * 8 + r) * N + n0 + t * 16 + nn] = acc1[t][r];
    }
}

// SuRoPE + repack: qkv fp32 [L][9216] -> Q,K bf16 [H][L][HD], V^T bf16 [H][HD][L].
__global__ __launch_bounds__(96) void rope_pack(
    const float* __restrict__ qkv, const float* __restrict__ sf,
    __bf16* __restrict__ Q, __bf16* __restrict__ Kc, __bf16* __restrict__ Vt) {
  const int d = threadIdx.x;
  const int h = blockIdx.x & (NHEAD - 1);
  const int l = blockIdx.x >> 5;
  const int dr = (d < 48) ? d : d - 48;
  const float invf = 1.0f / (sf[dr] * __powf(THETA, (float)(2 * dr) / (float)HDIM));
  const float ang = (float)l * invf;
  const float c = cosf(ang) * ROPE_SCALE;
  const float s = sinf(ang) * ROPE_SCALE;

  const size_t rowb = (size_t)l * OPQKV + (size_t)h * HDIM;
  const int dother = (d < 48) ? d + 48 : d - 48;
  const float q  = qkv[rowb + d];
  const float qo = qkv[rowb + dother];
  const float k  = qkv[rowb + OPROJ + d];
  const float ko = qkv[rowb + OPROJ + dother];
  const float v  = qkv[rowb + 2 * OPROJ + d];
  const float qr = (d < 48) ? (q * c - qo * s) : (q * c + qo * s);
  const float kr = (d < 48) ? (k * c - ko * s) : (k * c + ko * s);

  const size_t ho = ((size_t)h * LSEQ + l) * HDIM + d;
  Q[ho]  = (__bf16)qr;
  Kc[ho] = (__bf16)kr;
  Vt[((size_t)h * HDIM + d) * LSEQ + l] = (__bf16)v;
}

// Causal flash attention. One wave owns a 16-row q tile; loops over 32-key chunks.
// S = Q*K^T (6 wmma), online softmax in C-layout (rows = hf*8+r), P re-shuffled
// C-layout -> A-layout through a per-wave LDS tile, PV (6 wmma). V fragments are
// fetched right after the S wmmas so global latency hides under softmax VALU.
__global__ __launch_bounds__(128) void attn_causal(
    const __bf16* __restrict__ Q, const __bf16* __restrict__ Kc,
    const __bf16* __restrict__ Vt, __bf16* __restrict__ O) {
  __shared__ __align__(16) __bf16 Pl[4][16 * 32];
  const int lane = threadIdx.x & 31;
  const int wave = threadIdx.x >> 5;
  const int tile = blockIdx.x * 4 + wave;
  const int h  = tile >> 7;          // 128 q-tiles per head
  const int q0 = (tile & 127) << 4;
  const int hf = lane >> 4, nn = lane & 15;

  const __bf16* Qh = Q  + (size_t)h * LSEQ * HDIM;
  const __bf16* Kh = Kc + (size_t)h * LSEQ * HDIM;
  const __bf16* Vh = Vt + (size_t)h * HDIM * LSEQ;

  v16bf qa[3];
#pragma unroll
  for (int kk = 0; kk < 3; ++kk)
    qa[kk] = load_a_frag(Qh + (size_t)q0 * HDIM + kk * 32, HDIM, lane);

  float m_i[8], l_i[8];
  v8f o_acc[6] = {};
#pragma unroll
  for (int r = 0; r < 8; ++r) { m_i[r] = NEG_BIG; l_i[r] = 0.0f; }

  const int nk = (q0 + 16 + 31) >> 5;  // causal: keys [0, q0+16)
  for (int kc = 0; kc < nk; ++kc) {
    const int kb = kc << 5;
    v8f s0 = {}, s1 = {};
    {
      v16bf kb0[3], kb1[3];
#pragma unroll
      for (int kk = 0; kk < 3; ++kk) {   // batch loads, then wmmas
        kb0[kk] = load_b_frag(Kh + (size_t)kb * HDIM + kk * 32, HDIM, lane);
        kb1[kk] = load_b_frag(Kh + (size_t)(kb + 16) * HDIM + kk * 32, HDIM, lane);
      }
#pragma unroll
      for (int kk = 0; kk < 3; ++kk) {   // K-dim = HD = 96 = 3 x 32
        s0 = __builtin_amdgcn_wmma_f32_16x16x32_bf16(false, qa[kk], false, kb0[kk], (short)0, s0, false, false);
        s1 = __builtin_amdgcn_wmma_f32_16x16x32_bf16(false, qa[kk], false, kb1[kk], (short)0, s1, false, false);
      }
    }

    // Prefetch V fragments for this chunk; latency hides under softmax below.
    v16bf vb[6];
#pragma unroll
    for (int t = 0; t < 6; ++t)
      vb[t] = load_b_frag(Vh + (size_t)(t * 16) * LSEQ + kb, LSEQ, lane);

    float mloc[8], rs[8], alpha[8];
#pragma unroll
    for (int r = 0; r < 8; ++r) {
      const int qg = q0 + hf * 8 + r;
      float a0 = s0[r] * ATTN_SCALE;
      float a1 = s1[r] * ATTN_SCALE;
      if (kb + nn > qg)      a0 = NEG_BIG;   // causal mask
      if (kb + 16 + nn > qg) a1 = NEG_BIG;
      s0[r] = a0; s1[r] = a1;
      mloc[r] = fmaxf(a0, a1);
    }
#pragma unroll
    for (int off = 1; off < 16; off <<= 1)   // row-max across 16-lane halves
#pragma unroll
      for (int r = 0; r < 8; ++r)
        mloc[r] = fmaxf(mloc[r], __shfl_xor(mloc[r], off, 32));
#pragma unroll
    for (int r = 0; r < 8; ++r) {
      const float mn = fmaxf(m_i[r], mloc[r]);
      alpha[r] = __expf(m_i[r] - mn);
      m_i[r] = mn;
      const float p0 = __expf(s0[r] - mn);
      const float p1 = __expf(s1[r] - mn);
      Pl[wave][(hf * 8 + r) * 32 + nn]      = (__bf16)p0;
      Pl[wave][(hf * 8 + r) * 32 + 16 + nn] = (__bf16)p1;
      rs[r] = p0 + p1;
    }
#pragma unroll
    for (int off = 1; off < 16; off <<= 1)   // row-sum
#pragma unroll
      for (int r = 0; r < 8; ++r)
        rs[r] += __shfl_xor(rs[r], off, 32);
#pragma unroll
    for (int r = 0; r < 8; ++r) l_i[r] = l_i[r] * alpha[r] + rs[r];
#pragma unroll
    for (int t = 0; t < 6; ++t)
#pragma unroll
      for (int r = 0; r < 8; ++r) o_acc[t][r] *= alpha[r];

    // Intra-wave LDS RAW fence: P staging store/load are type-punned, keep them ordered.
    asm volatile("s_wait_dscnt 0" ::: "memory");
    v16bf pf = load_a_frag(&Pl[wave][0], 32, lane);
#pragma unroll
    for (int t = 0; t < 6; ++t)                    // N = HD = 96 = 6 x 16
      o_acc[t] = __builtin_amdgcn_wmma_f32_16x16x32_bf16(
          false, pf, false, vb[t], (short)0, o_acc[t], false, false);
  }

#pragma unroll
  for (int t = 0; t < 6; ++t)
#pragma unroll
    for (int r = 0; r < 8; ++r) {
      const float ov = o_acc[t][r] / l_i[r];
      O[(size_t)(q0 + hf * 8 + r) * OPROJ + h * HDIM + t * 16 + nn] = (__bf16)ov;
    }
}

extern "C" void kernel_launch(void* const* d_in, const int* in_sizes, int n_in,
                              void* d_out, int out_size, void* d_ws, size_t ws_size,
                              hipStream_t stream) {
  (void)in_sizes; (void)n_in; (void)out_size; (void)ws_size;
  const float* x    = (const float*)d_in[0];   // [L, D]
  const float* wqkv = (const float*)d_in[1];   // [OPQKV, D]
  const float* wo   = (const float*)d_in[2];   // [D, OPROJ]
  const float* sf   = (const float*)d_in[3];   // [HD/2]
  float* out = (float*)d_out;                  // [L, D]

  char* ws = (char*)d_ws;
  size_t off = 0;
  auto alloc = [&](size_t bytes) -> void* {
    void* p = (void*)(ws + off);
    off += (bytes + 255) & ~(size_t)255;
    return p;
  };
  // ~214 MB total workspace
  __bf16* xb  = (__bf16*)alloc((size_t)LSEQ * DMODEL * 2);
  __bf16* wqb = (__bf16*)alloc((size_t)OPQKV * DMODEL * 2);
  __bf16* wob = (__bf16*)alloc((size_t)DMODEL * OPROJ * 2);
  float*  qkv = (float*) alloc((size_t)LSEQ * OPQKV * 4);
  __bf16* Qb  = (__bf16*)alloc((size_t)NHEAD * LSEQ * HDIM * 2);
  __bf16* Kb  = (__bf16*)alloc((size_t)NHEAD * LSEQ * HDIM * 2);
  __bf16* Vt  = (__bf16*)alloc((size_t)NHEAD * HDIM * LSEQ * 2);
  __bf16* Ob  = (__bf16*)alloc((size_t)LSEQ * OPROJ * 2);

  cast_f32_bf16<<<1024, 256, 0, stream>>>(x,    xb,  (long)LSEQ * DMODEL);
  cast_f32_bf16<<<2048, 256, 0, stream>>>(wqkv, wqb, (long)OPQKV * DMODEL);
  cast_f32_bf16<<<1024, 256, 0, stream>>>(wo,   wob, (long)DMODEL * OPROJ);

  dim3 g1(OPQKV / 64, LSEQ / 128);
  gemm_bf16_f32<<<g1, 128, 0, stream>>>(xb, wqb, qkv, LSEQ, OPQKV, DMODEL);

  rope_pack<<<LSEQ * NHEAD, 96, 0, stream>>>(qkv, sf, Qb, Kb, Vt);

  attn_causal<<<NHEAD * (LSEQ / 16) / 4, 128, 0, stream>>>(Qb, Kb, Vt, Ob);

  dim3 g2(DMODEL / 64, LSEQ / 128);
  gemm_bf16_f32<<<g2, 128, 0, stream>>>(Ob, wob, out, LSEQ, DMODEL, OPROJ);
}
